// MixtureOfNonlinearExpertsPolicy_29300266893926
// MI455X (gfx1250) — matrix operand order repaired and checked by reading, not compiled, not run on hardware
//
#include <hip/hip_runtime.h>
#include <hip/hip_bf16.h>
#include <stdint.h>
#include <stddef.h>

// Problem constants (match reference)
#define B_TOTAL   32768
#define DIM_X     511
#define DIM_IN    512
#define H_EXP     384
#define H_GATE    264
#define H_GATE_P  320     // padded: 20 n-tiles of 16 (5 per wave), 10 k-steps of 32
#define NUM_E     16
#define DIM_U     256

// Workspace layout (bytes), 1KB-fragment aligned
#define OFF_GW1   0u                       // 320 frags * 1KB = 327680
#define OFF_GW2   327680u                  // 10 frags       -> 337920
#define OFF_GB1   337920u                  // 320 floats     -> 339200
#define OFF_W1    339968u                  // 6144 frags     -> 6631424
#define OFF_W2    6631424u                 // 3072 frags     -> 9777152 (~9.8MB)

typedef __bf16        v16bf __attribute__((ext_vector_type(16)));
typedef float         v8f   __attribute__((ext_vector_type(8)));
typedef unsigned int  v4u   __attribute__((ext_vector_type(4)));
typedef float         v4f   __attribute__((ext_vector_type(4)));

union FragU {
  v16bf v;
  v4u   q[2];
  unsigned short u[16];
};

__device__ __forceinline__ unsigned short f2bf(float f) {
  unsigned int u = __float_as_uint(f);
  u += 0x7fffu + ((u >> 16) & 1u);        // round-to-nearest-even
  return (unsigned short)(u >> 16);
}

__device__ __forceinline__ float fast_tanh(float x) {
  x = fminf(fmaxf(x, -15.f), 15.f);
  float e = __expf(2.f * x);
  return (e - 1.f) / (e + 1.f);
}

__device__ __forceinline__ v8f wmma_bf16(v16bf a, v16bf b, v8f c) {
  // D = A(16x32 bf16) * B(32x16 bf16) + C(16x16 f32)
  return __builtin_amdgcn_wmma_f32_16x16x32_bf16(false, a, false, b,
                                                 (short)0, c, false, false);
}

// A fragment from a bf16 row in LDS. lanes 0-15: M=lane, K={k0..k0+7, k0+16..23};
// lanes 16-31: M=lane-16, K offset +8 (per 16-bit A layout).
__device__ __forceinline__ v16bf load_a_lds(const unsigned short* row, int k0, int koff) {
  FragU f;
  f.q[0] = *(const v4u*)(row + k0 + koff);
  f.q[1] = *(const v4u*)(row + k0 + 16 + koff);
  return f.v;
}

// B fragment pre-packed in workspace: frag = 32 lanes x 32B contiguous.
__device__ __forceinline__ v16bf load_b_ws(const char* base) {
  FragU f;
  f.q[0] = *(const v4u*)(base);
  f.q[1] = *(const v4u*)(base + 16);
  return f.v;
}

__device__ __forceinline__ v8f zero8() {
  v8f z = {0.f, 0.f, 0.f, 0.f, 0.f, 0.f, 0.f, 0.f};
  return z;
}

// ---------------------------------------------------------------------------
// Kernel 1: pack weights f32 -> bf16 WMMA B-fragment layout (one wave / frag)
// B-frag (K=32): lanes 0-15: elem j = W[k0+j+8*(j/8)][n0+lane]; lanes 16-31: W[k0+8+j+8*(j/8)][n0+lane-16]
// ids: [0,6144) W1 | [6144,9216) W2 | [9216,9536) gw1 | [9536,9546) gw2 | 9546 gb1
// ---------------------------------------------------------------------------
__global__ void prep_pack(const float* __restrict__ gw1, const float* __restrict__ gb1,
                          const float* __restrict__ gw2, const float* __restrict__ W1,
                          const float* __restrict__ W2, char* __restrict__ ws)
{
  int w = blockIdx.x * 4 + (threadIdx.x >> 5);
  int lane = threadIdx.x & 31;

  if (w == 9546) {  // padded gb1
    float* dst = (float*)(ws + OFF_GB1);
    for (int i = lane; i < H_GATE_P; i += 32) dst[i] = (i < H_GATE) ? gb1[i] : 0.f;
    return;
  }
  if (w > 9546) return;

  const float* src; int nstride, k0, n0; char* dst;
  int kmax = 1 << 30, nmax = 1 << 30;
  if (w < 6144) {                       // W1: id = e*384 + kt*24 + nt
    int e = w / 384, rem = w % 384, kt = rem / 24, nt = rem % 24;
    src = W1 + (size_t)e * DIM_IN * H_EXP; nstride = H_EXP;
    k0 = kt * 32; n0 = nt * 16;
    dst = ws + OFF_W1 + (size_t)w * 1024;
  } else if (w < 9216) {                // W2: id = e*192 + kt*16 + nt
    int w2 = w - 6144;
    int e = w2 / 192, rem = w2 % 192, kt = rem / 16, nt = rem % 16;
    src = W2 + (size_t)e * H_EXP * DIM_U; nstride = DIM_U;
    k0 = kt * 32; n0 = nt * 16;
    dst = ws + OFF_W2 + (size_t)w2 * 1024;
  } else if (w < 9536) {                // gw1 padded: id = kt*20 + nt
    int wg = w - 9216; int kt = wg / 20, nt = wg % 20;
    src = gw1; nstride = H_GATE; nmax = H_GATE;
    k0 = kt * 32; n0 = nt * 16;
    dst = ws + OFF_GW1 + (size_t)wg * 1024;
  } else {                              // gw2 padded: id = kt (10 k-steps)
    int kt = w - 9536;
    src = gw2; nstride = NUM_E; kmax = H_GATE;
    k0 = kt * 32; n0 = 0;
    dst = ws + OFF_GW2 + (size_t)kt * 1024;
  }

  int kb = k0 + ((lane >= 16) ? 8 : 0);
  int n  = n0 + (lane & 15);
  FragU f;
#pragma unroll
  for (int j = 0; j < 16; ++j) {
    int k = kb + j + ((j >= 8) ? 8 : 0);
    float v = (k < kmax && n < nmax) ? src[(size_t)k * nstride + n] : 0.f;
    f.u[j] = f2bf(v);
  }
  *(v4u*)(dst + lane * 32)      = f.q[0];
  *(v4u*)(dst + lane * 32 + 16) = f.q[1];
}

// ---------------------------------------------------------------------------
// Kernel 2: gating net -> p  (16 batch rows / workgroup, 4 waves)
// kt-outer / persistent accumulators: A fragment loaded once per k-step and
// consumed immediately (no long-lived A sets -> no spills).
// ---------------------------------------------------------------------------
__global__ void __launch_bounds__(128, 1)
gating_kernel(const float* __restrict__ t, const float* __restrict__ x,
              const float* __restrict__ gb2, const char* __restrict__ ws,
              float* __restrict__ p_out)
{
  __shared__ __align__(16) unsigned short s_inp[16 * DIM_IN];
  __shared__ __align__(16) unsigned short s_g1[16 * H_GATE_P];
  __shared__ float s_log[16 * 16];

  int tid  = threadIdx.x;
  int lane = tid & 31, wv = tid >> 5;
  int b0   = blockIdx.x * 16;

  for (int idx = tid; idx < 16 * DIM_IN; idx += 128) {
    int row = idx >> 9, col = idx & 511;
    float v = (col == 0) ? t[b0 + row] : x[(size_t)(b0 + row) * DIM_X + col - 1];
    s_inp[idx] = f2bf(v);
  }
  __syncthreads();

  const float* gb1p = (const float*)(ws + OFF_GB1);
  int N = lane & 15, half8 = (lane >> 4) << 3;
  int koff = (lane >= 16) ? 8 : 0;
  const unsigned short* arow = s_inp + (lane & 15) * DIM_IN;

  // layer 1: inp[16x512] @ gw1[512x320] -> tanh -> s_g1 ; wave owns nt = wv*5..wv*5+4
  {
    v8f accs[5];
#pragma unroll
    for (int i = 0; i < 5; ++i) accs[i] = zero8();
    for (int kt = 0; kt < 16; ++kt) {
      v16bf a = load_a_lds(arow, kt * 32, koff);
      const char* wb = ws + OFF_GW1 + (size_t)(kt * 20 + wv * 5) * 1024 + lane * 32;
      __builtin_prefetch(wb + 20 * 1024, 0, 3);   // next k-step's slab
#pragma unroll
      for (int i = 0; i < 5; ++i) {
        v16bf b = load_b_ws(wb + (size_t)i * 1024);
        accs[i] = wmma_bf16(a, b, accs[i]);
      }
    }
#pragma unroll
    for (int i = 0; i < 5; ++i) {
      int n = (wv * 5 + i) * 16 + N;
      float bias = gb1p[n];
#pragma unroll
      for (int r = 0; r < 8; ++r) {
        float v = fast_tanh(accs[i][r] + bias);
        s_g1[(r + half8) * H_GATE_P + n] = f2bf(v);
      }
    }
  }
  __syncthreads();

  // layer 2: g1[16x320] @ gw2[320x16] -> logits (wave 0 only; 10 WMMAs)
  if (wv == 0) {
    v8f acc = zero8();
    const unsigned short* arow2 = s_g1 + (lane & 15) * H_GATE_P;
    for (int kt = 0; kt < 10; ++kt) {
      v16bf a = load_a_lds(arow2, kt * 32, koff);
      v16bf b = load_b_ws(ws + OFF_GW2 + (size_t)kt * 1024 + lane * 32);
      acc = wmma_bf16(a, b, acc);
    }
    float bias = gb2[N];
#pragma unroll
    for (int r = 0; r < 8; ++r) s_log[(r + half8) * 16 + N] = acc[r] + bias;
  }
  __syncthreads();

  // softmax: one thread per row
  if (tid < 16) {
    float l[16], m = -1e30f;
#pragma unroll
    for (int e = 0; e < 16; ++e) { l[e] = s_log[tid * 16 + e]; m = fmaxf(m, l[e]); }
    float s = 0.f;
#pragma unroll
    for (int e = 0; e < 16; ++e) { l[e] = __expf(l[e] - m); s += l[e]; }
    float inv = 1.f / s;
    float* pr = p_out + (size_t)(b0 + tid) * NUM_E;
#pragma unroll
    for (int e = 0; e < 16; ++e) pr[e] = l[e] * inv;
  }
}

// ---------------------------------------------------------------------------
// Kernel 3: all 16 experts for a 16-row batch tile; u accumulation; U staged
// in LDS per 8-expert group so global U writes are 32B-contiguous.
// Dynamic LDS: s_U 128KB | s_p 1KB | s_inp 16KB | s_h 12KB  = 157KB
// ---------------------------------------------------------------------------
#define SM_U_OFF    0
#define SM_P_OFF    131072
#define SM_INP_OFF  132096
#define SM_H_OFF    148480
#define SM_TOTAL    160768

__global__ void __launch_bounds__(128, 1)
expert_kernel(const float* __restrict__ t, const float* __restrict__ x,
              const float* __restrict__ b1, const float* __restrict__ b2,
              const char* __restrict__ ws, const float* __restrict__ p_in,
              float* __restrict__ u_out, float* __restrict__ U_out)
{
  extern __shared__ char smem[];
  float*          s_U   = (float*)(smem + SM_U_OFF);            // [16][256][8]
  float*          s_p   = (float*)(smem + SM_P_OFF);            // [16][16]
  unsigned short* s_inp = (unsigned short*)(smem + SM_INP_OFF); // [16][512] bf16
  unsigned short* s_h   = (unsigned short*)(smem + SM_H_OFF);   // [16][384] bf16

  int tid  = threadIdx.x;
  int lane = tid & 31, wv = tid >> 5;   // wv = N-quarter owned by this wave
  int b0   = blockIdx.x * 16;

  for (int idx = tid; idx < 16 * DIM_IN; idx += 128) {
    int row = idx >> 9, col = idx & 511;
    float v = (col == 0) ? t[b0 + row] : x[(size_t)(b0 + row) * DIM_X + col - 1];
    s_inp[idx] = f2bf(v);
  }
  for (int idx = tid; idx < 256; idx += 128)
    s_p[idx] = p_in[(size_t)b0 * NUM_E + idx];
  __syncthreads();

  int N = lane & 15, half8 = (lane >> 4) << 3;
  int koff = (lane >= 16) ? 8 : 0;
  const unsigned short* arow_inp = s_inp + (lane & 15) * DIM_IN;
  const unsigned short* arow_h   = s_h   + (lane & 15) * H_EXP;

  v8f u_acc[4];
#pragma unroll
  for (int j = 0; j < 4; ++j) u_acc[j] = zero8();

  for (int e = 0; e < NUM_E; ++e) {
    // ---- layer 1: inp @ W1[e] over 6 N-tiles, 16 k-steps (A frag shared)
    v8f c1[6];
#pragma unroll
    for (int j = 0; j < 6; ++j) c1[j] = zero8();
    for (int kt = 0; kt < 16; ++kt) {
      v16bf a = load_a_lds(arow_inp, kt * 32, koff);
      const char* wb = ws + OFF_W1 + (size_t)((e * 16 + kt) * 24 + wv * 6) * 1024 + lane * 32;
      __builtin_prefetch(wb + 24 * 1024, 0, 3);   // next k-step's slab
#pragma unroll
      for (int j = 0; j < 6; ++j) {
        v16bf b = load_b_ws(wb + (size_t)j * 1024);
        c1[j] = wmma_bf16(a, b, c1[j]);
      }
    }
    __syncthreads();   // previous expert's layer-2 reads of s_h complete
    // ---- bias + tanh -> s_h (bf16)
#pragma unroll
    for (int j = 0; j < 6; ++j) {
      int n = (wv * 6 + j) * 16 + N;
      float bias = b1[e * H_EXP + n];
#pragma unroll
      for (int r = 0; r < 8; ++r) {
        float v = fast_tanh(c1[j][r] + bias);
        s_h[(r + half8) * H_EXP + n] = f2bf(v);
      }
    }
    __syncthreads();   // s_h ready

    // ---- layer 2: h @ W2[e] over 4 N-tiles, 12 k-steps
    v8f c2[4];
#pragma unroll
    for (int j = 0; j < 4; ++j) c2[j] = zero8();
    for (int kt = 0; kt < 12; ++kt) {
      v16bf a = load_a_lds(arow_h, kt * 32, koff);
      const char* wb = ws + OFF_W2 + (size_t)((e * 12 + kt) * 16 + wv * 4) * 1024 + lane * 32;
      __builtin_prefetch(wb + 16 * 1024, 0, 3);   // next k-step's slab
#pragma unroll
      for (int j = 0; j < 4; ++j) {
        v16bf b = load_b_ws(wb + (size_t)j * 1024);
        c2[j] = wmma_bf16(a, b, c2[j]);
      }
    }

    // ---- bias, accumulate u, stage Ue into s_U (e-contiguous, groups of 8)
    float pv[8];
#pragma unroll
    for (int r = 0; r < 8; ++r) pv[r] = s_p[(r + half8) * NUM_E + e];
#pragma unroll
    for (int j = 0; j < 4; ++j) {
      int o = (wv * 4 + j) * 16 + N;
      float bias = b2[e * DIM_U + o];
#pragma unroll
      for (int r = 0; r < 8; ++r) {
        float v = c2[j][r] + bias;
        u_acc[j][r] += v * pv[r];
        s_U[(size_t)((r + half8) * DIM_U + o) * 8 + (e & 7)] = v;
      }
    }

    if ((e & 7) == 7) {
      __syncthreads();             // all waves' s_U writes done
      int ebase = e - 7;
      for (int i = tid; i < 16 * DIM_U; i += 128) {  // i = m*256 + o
        const float* srcp = s_U + (size_t)i * 8;
        v4f lo = *(const v4f*)(srcp);
        v4f hi = *(const v4f*)(srcp + 4);
        float* dstp = U_out + ((size_t)(b0 + (i >> 8)) * DIM_U + (i & 255)) * NUM_E + ebase;
        *(v4f*)dstp       = lo;    // 32B contiguous per (b,o)
        *(v4f*)(dstp + 4) = hi;
      }
      // next group's s_U writes occur only after the next iteration's barriers
    }
  }

  // ---- write u (coalesced per half-wave)
#pragma unroll
  for (int j = 0; j < 4; ++j) {
    int o = (wv * 4 + j) * 16 + N;
#pragma unroll
    for (int r = 0; r < 8; ++r)
      u_out[(size_t)(b0 + r + half8) * DIM_U + o] = u_acc[j][r];
  }
}

// ---------------------------------------------------------------------------
extern "C" void kernel_launch(void* const* d_in, const int* in_sizes, int n_in,
                              void* d_out, int out_size, void* d_ws, size_t ws_size,
                              hipStream_t stream) {
  const float* t   = (const float*)d_in[0];
  const float* x   = (const float*)d_in[1];
  const float* gw1 = (const float*)d_in[2];
  const float* gb1 = (const float*)d_in[3];
  const float* gw2 = (const float*)d_in[4];
  const float* gb2 = (const float*)d_in[5];
  const float* W1  = (const float*)d_in[6];
  const float* b1  = (const float*)d_in[7];
  const float* W2  = (const float*)d_in[8];
  const float* b2  = (const float*)d_in[9];

  float* out   = (float*)d_out;
  float* u_out = out;                                            // [B,256]
  float* p_out = out + (size_t)B_TOTAL * DIM_U;                  // [B,16]
  float* U_out = p_out + (size_t)B_TOTAL * NUM_E;                // [B,256,16]
  char*  ws    = (char*)d_ws;

  // 9547 fragment-waves, 4 waves / 128-thread block
  prep_pack<<<2387, 128, 0, stream>>>(gw1, gb1, gw2, W1, W2, ws);
  gating_kernel<<<B_TOTAL / 16, 128, 0, stream>>>(t, x, gb2, ws, p_out);
  expert_kernel<<<B_TOTAL / 16, 128, SM_TOTAL, stream>>>(t, x, b1, b2, ws, p_out,
                                                         u_out, U_out);
}
